// LocalCausalAttention_9320079033003
// MI455X (gfx1250) — matrix operand order repaired and checked by reading, not compiled, not run on hardware
//
#include <hip/hip_runtime.h>

#define DMODEL 1024
#define NHEADS 16
#define DKH    64
#define BATCH  2
#define TSEQ   2048
#define WIN    256
#define BT     (BATCH*TSEQ)      // 4096 rows

typedef __bf16 bf16_t;
typedef __attribute__((ext_vector_type(16))) __bf16 v16bf;
typedef __attribute__((ext_vector_type(8)))  __bf16 v8bf;
typedef __attribute__((ext_vector_type(8)))  float  v8f;

__device__ __forceinline__ bf16_t f2bf(float f) { return (bf16_t)f; }

// combine two contiguous 8x bf16 (16B) loads into one A/B operand vector
__device__ __forceinline__ v16bf ld16(const bf16_t* p0, const bf16_t* p1) {
    v8bf lo = *(const v8bf*)p0;
    v8bf hi = *(const v8bf*)p1;
    v16bf r;
    #pragma unroll
    for (int e = 0; e < 8; e++) { r[e] = lo[e]; r[e + 8] = hi[e]; }
    return r;
}
__device__ __forceinline__ v16bf zero16() {
    v16bf z;
    #pragma unroll
    for (int e = 0; e < 16; e++) z[e] = (bf16_t)0.0f;
    return z;
}

__device__ __forceinline__ float grpmax16(float v) {
    v = fmaxf(v, __shfl_xor(v, 1, 32));
    v = fmaxf(v, __shfl_xor(v, 2, 32));
    v = fmaxf(v, __shfl_xor(v, 4, 32));
    v = fmaxf(v, __shfl_xor(v, 8, 32));
    return v;
}
__device__ __forceinline__ float grpsum16(float v) {
    v += __shfl_xor(v, 1, 32);
    v += __shfl_xor(v, 2, 32);
    v += __shfl_xor(v, 4, 32);
    v += __shfl_xor(v, 8, 32);
    return v;
}

// ---------------- LayerNorm: one wave per row, bf16 output ----------------
__global__ __launch_bounds__(256) void ln_kernel(
    const float* __restrict__ x, const float* __restrict__ gamma,
    const float* __restrict__ beta, bf16_t* __restrict__ hbf)
{
    int row  = blockIdx.x * 8 + (threadIdx.x >> 5);
    int lane = threadIdx.x & 31;
    const float* xr = x + (size_t)row * DMODEL;

    float4 v[8];
    float s = 0.f, s2 = 0.f;
    #pragma unroll
    for (int c = 0; c < 8; c++) {
        v[c] = ((const float4*)xr)[lane + c * 32];
        s  += v[c].x + v[c].y + v[c].z + v[c].w;
        s2 += v[c].x*v[c].x + v[c].y*v[c].y + v[c].z*v[c].z + v[c].w*v[c].w;
    }
    #pragma unroll
    for (int m = 1; m < 32; m <<= 1) {
        s  += __shfl_xor(s,  m, 32);
        s2 += __shfl_xor(s2, m, 32);
    }
    float mu   = s * (1.0f / DMODEL);
    float var  = s2 * (1.0f / DMODEL) - mu * mu;
    float rstd = rsqrtf(var + 1e-5f);

    bf16_t* hr = hbf + (size_t)row * DMODEL;
    #pragma unroll
    for (int c = 0; c < 8; c++) {
        float vv[4]; *(float4*)vv = v[c];
        int base = (lane + c * 32) * 4;
        #pragma unroll
        for (int e = 0; e < 4; e++) {
            int col = base + e;
            hr[col] = f2bf((vv[e] - mu) * rstd * gamma[col] + beta[col]);
        }
    }
}

// ------------- weight convert + transpose: Wt[n*D + k] = bf16(W[k*D + n]) ----
__global__ __launch_bounds__(256) void wconv_kernel(
    const float* __restrict__ W, bf16_t* __restrict__ Wt)
{
    int idx = blockIdx.x * 256 + threadIdx.x;       // coalesced read
    int k = idx / DMODEL, n = idx % DMODEL;
    Wt[(size_t)n * DMODEL + k] = f2bf(W[idx]);
}

// ------------- V transpose: Vt[((b*H+h)*dk + d)*T + t] -----------------------
__global__ __launch_bounds__(256) void vtrans_kernel(
    const bf16_t* __restrict__ Vbf, bf16_t* __restrict__ Vt)
{
    int idx = blockIdx.x * 256 + threadIdx.x;       // over B*T*D
    int bt = idx / DMODEL, d = idx % DMODEL;
    int b = bt / TSEQ, t = bt % TSEQ;
    int h = d / DKH, dd = d % DKH;
    Vt[(((size_t)b * NHEADS + h) * DKH + dd) * TSEQ + t] = Vbf[idx];
}

// ---- 16x64 bf16 WMMA tile, ping-pong software pipeline (no register copies)
__device__ __forceinline__ void gemm_tile_16x64(
    const bf16_t* __restrict__ A, const bf16_t* __restrict__ Bt,
    int m0, int n0, int Kdim, v8f acc[4])
{
    int lane = threadIdx.x & 31;
    int half = lane >> 4, id = lane & 15;
    int sel8 = half * 8, sel16 = half * 16;
    const bf16_t* arow = A + (size_t)(m0 + id) * Kdim;
    const bf16_t* bcol[4];
    #pragma unroll
    for (int nt = 0; nt < 4; nt++)
        bcol[nt] = Bt + (size_t)(n0 + nt * 16 + id) * Kdim + sel16;

    // prologue: load k-step 0 into buffer 0
    v16bf a0 = ld16(arow + sel16, arow + sel16 + 8);
    v16bf b0[4];
    #pragma unroll
    for (int nt = 0; nt < 4; nt++)
        b0[nt] = ld16(bcol[nt], bcol[nt] + 8);

    // Kdim is a multiple of 64: two k-steps per iteration, buffers alternate
    for (int k0 = 0; k0 < Kdim; k0 += 64) {
        int k1 = k0 + 32;
        v16bf a1 = ld16(arow + k1 + sel16, arow + k1 + sel16 + 8);
        v16bf b1[4];
        #pragma unroll
        for (int nt = 0; nt < 4; nt++)
            b1[nt] = ld16(bcol[nt] + k1, bcol[nt] + k1 + 8);
        #pragma unroll
        for (int nt = 0; nt < 4; nt++)
            acc[nt] = __builtin_amdgcn_wmma_f32_16x16x32_bf16(
                false, a0, false, b0[nt], (short)0, acc[nt], false, false);

        int k2 = k0 + 64;
        if (k2 < Kdim) {
            a0 = ld16(arow + k2 + sel16, arow + k2 + sel16 + 8);
            #pragma unroll
            for (int nt = 0; nt < 4; nt++)
                b0[nt] = ld16(bcol[nt] + k2, bcol[nt] + k2 + 8);
        }
        #pragma unroll
        for (int nt = 0; nt < 4; nt++)
            acc[nt] = __builtin_amdgcn_wmma_f32_16x16x32_bf16(
                false, a1, false, b1[nt], (short)0, acc[nt], false, false);
    }
}

// ------------- QKV GEMM: blockIdx.y selects Wq/Wk/Wv, bf16 out --------------
__global__ __launch_bounds__(256) void gemm_qkv_kernel(
    const bf16_t* __restrict__ hbf,
    const bf16_t* __restrict__ WtQ, const bf16_t* __restrict__ WtK,
    const bf16_t* __restrict__ WtV,
    bf16_t* __restrict__ Q, bf16_t* __restrict__ K, bf16_t* __restrict__ V)
{
    int wid = blockIdx.x * 8 + (threadIdx.x >> 5);   // 0..4095
    int mt = wid >> 4, ng = wid & 15;
    const bf16_t* Wt = (blockIdx.y == 0) ? WtQ : (blockIdx.y == 1) ? WtK : WtV;
    bf16_t* dst      = (blockIdx.y == 0) ? Q   : (blockIdx.y == 1) ? K   : V;

    v8f acc[4];
    #pragma unroll
    for (int i = 0; i < 4; i++)
        #pragma unroll
        for (int e = 0; e < 8; e++) acc[i][e] = 0.f;

    gemm_tile_16x64(hbf, Wt, mt * 16, ng * 64, DMODEL, acc);

    int lane = threadIdx.x & 31;
    int half = lane >> 4, id = lane & 15;
    #pragma unroll
    for (int nt = 0; nt < 4; nt++)
        #pragma unroll
        for (int r = 0; r < 8; r++) {
            int m = mt * 16 + r + half * 8;
            int n = ng * 64 + nt * 16 + id;
            dst[(size_t)m * DMODEL + n] = f2bf(acc[nt][r]);
        }
}

// ------------- output GEMM: out = x + O @ Wo + bo (fp32) --------------------
__global__ __launch_bounds__(256) void gemm_out_kernel(
    const bf16_t* __restrict__ Obf, const bf16_t* __restrict__ WtO,
    const float* __restrict__ x, const float* __restrict__ bo,
    float* __restrict__ out)
{
    int wid = blockIdx.x * 8 + (threadIdx.x >> 5);
    int mt = wid >> 4, ng = wid & 15;

    v8f acc[4];
    #pragma unroll
    for (int i = 0; i < 4; i++)
        #pragma unroll
        for (int e = 0; e < 8; e++) acc[i][e] = 0.f;

    gemm_tile_16x64(Obf, WtO, mt * 16, ng * 64, DMODEL, acc);

    int lane = threadIdx.x & 31;
    int half = lane >> 4, id = lane & 15;
    #pragma unroll
    for (int nt = 0; nt < 4; nt++)
        #pragma unroll
        for (int r = 0; r < 8; r++) {
            int m = mt * 16 + r + half * 8;
            int n = ng * 64 + nt * 16 + id;
            size_t idx = (size_t)m * DMODEL + n;
            out[idx] = x[idx] + acc[nt][r] + bo[n];
        }
}

// ------------- flash attention, sliding causal window, wave per 16-q tile ---
__global__ __launch_bounds__(256) void attn_kernel(
    const bf16_t* __restrict__ Q, const bf16_t* __restrict__ Kb,
    const bf16_t* __restrict__ Vt, bf16_t* __restrict__ O)
{
    __shared__ __attribute__((aligned(16))) bf16_t pbuf[8][16 * 32];

    int w    = threadIdx.x >> 5;
    int lane = threadIdx.x & 31;
    int half = lane >> 4, id = lane & 15;
    int sel8 = half * 8, sel16 = half * 16;

    int wid = blockIdx.x * 8 + w;          // 0..4095
    int qt  = wid & 127;                   // query tile (T/16 = 128)
    int bh  = wid >> 7;                    // 0..31
    int b   = bh >> 4, h = bh & 15;
    int i0  = qt * 16;
    size_t tokbase = (size_t)b * TSEQ;
    int hoff = h * DKH;

    // Q as two A operands (dk contraction halves 0..31, 32..63), row = id
    const bf16_t* qrow = Q + (tokbase + i0 + id) * DMODEL + hoff;
    v16bf qa[2];
    #pragma unroll
    for (int kk = 0; kk < 2; kk++)
        qa[kk] = ld16(qrow + kk * 32 + sel16, qrow + kk * 32 + sel16 + 8);

    v8f oacc[4];
    #pragma unroll
    for (int i = 0; i < 4; i++)
        #pragma unroll
        for (int e = 0; e < 8; e++) oacc[i][e] = 0.f;

    float mrow[8], lrow[8];
    #pragma unroll
    for (int r = 0; r < 8; r++) { mrow[r] = -3.0e38f; lrow[r] = 0.f; }

    const float scale = 0.125f;            // 1/sqrt(64)
    const bf16_t* Kbase = Kb + tokbase * DMODEL + hoff;
    const bf16_t* Vbase = Vt + (size_t)bh * DKH * TSEQ;

    // K-tile loader for chunk ct (32 keys = 2 tiles x 2 dk-halves)
    auto loadK = [&](int ct, v16bf (&kb)[2][2]) {
        int jb = (qt - 16 + 2 * ct) * 16;
        #pragma unroll
        for (int ts = 0; ts < 2; ts++) {
            int tok = jb + ts * 16 + id;               // key token for this column
            bool ok = (tok >= 0) && (tok < TSEQ);
            const bf16_t* kr = Kbase + (size_t)(ok ? tok : 0) * DMODEL;
            #pragma unroll
            for (int kk = 0; kk < 2; kk++)
                kb[ts][kk] = ok ? ld16(kr + kk * 32 + sel16, kr + kk * 32 + sel16 + 8)
                               : zero16();
        }
    };

    // one 32-key chunk: scores with kc, prefetch chunk ct+1 into kn, softmax, PV
    auto chunk = [&](int ct, v16bf (&kc)[2][2], v16bf (&kn)[2][2], bool pf) {
        int jbase = (qt - 16 + 2 * ct) * 16;

        v8f s[2];
        #pragma unroll
        for (int ts = 0; ts < 2; ts++)
            #pragma unroll
            for (int e = 0; e < 8; e++) s[ts][e] = 0.f;
        #pragma unroll
        for (int ts = 0; ts < 2; ts++)
            #pragma unroll
            for (int kk = 0; kk < 2; kk++)
                s[ts] = __builtin_amdgcn_wmma_f32_16x16x32_bf16(
                    false, qa[kk], false, kc[ts][kk], (short)0, s[ts], false, false);

        // prefetch next chunk's K tiles; latency hidden by softmax below
        if (pf) loadK(ct + 1, kn);

        // mask + scale (C layout: row = r + half*8, col = id)
        float sm[2][8];
        #pragma unroll
        for (int ts = 0; ts < 2; ts++)
            #pragma unroll
            for (int r = 0; r < 8; r++) {
                int i = i0 + r + half * 8;
                int j = jbase + ts * 16 + id;
                bool valid = (j >= 0) && (j <= i) && (j > i - WIN);
                sm[ts][r] = valid ? (s[ts][r] * scale) : -1e30f;
            }

        // online softmax update + stage P into LDS (C->A transpose)
        #pragma unroll
        for (int r = 0; r < 8; r++) {
            float t  = grpmax16(fmaxf(sm[0][r], sm[1][r]));
            float mn = fmaxf(mrow[r], t);
            float alpha = __expf(mrow[r] - mn);
            mrow[r] = mn;
            float p0 = (sm[0][r] > -1e29f) ? __expf(sm[0][r] - mn) : 0.f;
            float p1 = (sm[1][r] > -1e29f) ? __expf(sm[1][r] - mn) : 0.f;
            lrow[r] = lrow[r] * alpha + grpsum16(p0 + p1);
            #pragma unroll
            for (int nt = 0; nt < 4; nt++) oacc[nt][r] *= alpha;
            int mr = r + half * 8;
            pbuf[w][mr * 32 + id]      = f2bf(p0);
            pbuf[w][mr * 32 + 16 + id] = f2bf(p1);
        }
        asm volatile("s_wait_dscnt 0" ::: "memory");

        // reload P in A layout (16 rows x 32 keys)
        const bf16_t* pr = &pbuf[w][id * 32];
        v16bf pa = ld16(pr + sel16, pr + sel16 + 8);

        // O += P @ V : 4 d-chunks of 16, contraction over 32 keys
        #pragma unroll
        for (int dc = 0; dc < 4; dc++) {
            const bf16_t* vr = Vbase + (size_t)(dc * 16 + id) * TSEQ;
            v16bf vb;
            int js = jbase + sel16;
            if (js >= 0 && js + 15 < TSEQ) {
                vb = ld16(vr + js, vr + js + 8);
            } else {
                #pragma unroll
                for (int e = 0; e < 16; e++) {
                    int j = js + e;
                    vb[e] = (j >= 0 && j < TSEQ) ? vr[j] : (bf16_t)0.0f;
                }
            }
            oacc[dc] = __builtin_amdgcn_wmma_f32_16x16x32_bf16(
                false, pa, false, vb, (short)0, oacc[dc], false, false);
        }
    };

    // 9 chunks of 32 keys cover the 17-tile window; ping-pong K buffers
    v16bf kA[2][2], kB[2][2];
    loadK(0, kA);
    for (int c2 = 0; c2 < 4; c2++) {
        chunk(2 * c2,     kA, kB, true);   // uses A, prefetches into B
        chunk(2 * c2 + 1, kB, kA, true);   // uses B, prefetches into A
    }
    chunk(8, kA, kB, false);               // epilogue chunk, no prefetch

    // ---- normalize and store O (bf16 for the Wo GEMM)
    #pragma unroll
    for (int r = 0; r < 8; r++) {
        float inv = 1.0f / lrow[r];
        int m = i0 + r + half * 8;
        #pragma unroll
        for (int dc = 0; dc < 4; dc++)
            O[(tokbase + m) * DMODEL + hoff + dc * 16 + id] = f2bf(oacc[dc][r] * inv);
    }
}

// ---------------------------------------------------------------------------
extern "C" void kernel_launch(void* const* d_in, const int* in_sizes, int n_in,
                              void* d_out, int out_size, void* d_ws, size_t ws_size,
                              hipStream_t stream) {
    const float* x     = (const float*)d_in[0];
    const float* Wq    = (const float*)d_in[1];
    const float* Wk    = (const float*)d_in[2];
    const float* Wv    = (const float*)d_in[3];
    const float* Wo    = (const float*)d_in[4];
    const float* bo    = (const float*)d_in[5];
    const float* gamma = (const float*)d_in[6];
    const float* beta  = (const float*)d_in[7];
    float* out = (float*)d_out;

    const size_t NBTD = (size_t)BT * DMODEL;       // 4M elements
    const size_t NDD  = (size_t)DMODEL * DMODEL;   // 1M elements
    bf16_t* hbf = (bf16_t*)d_ws;
    bf16_t* wtq = hbf + NBTD;
    bf16_t* wtk = wtq + NDD;
    bf16_t* wtv = wtk + NDD;
    bf16_t* wto = wtv + NDD;
    bf16_t* qbf = wto + NDD;
    bf16_t* kbf = qbf + NBTD;
    bf16_t* vbf = kbf + NBTD;
    bf16_t* vtb = vbf + NBTD;
    bf16_t* obf = vtb + NBTD;                      // total 28M elems = 56 MB

    ln_kernel<<<BT / 8, 256, 0, stream>>>(x, gamma, beta, hbf);
    wconv_kernel<<<NDD / 256, 256, 0, stream>>>(Wq, wtq);
    wconv_kernel<<<NDD / 256, 256, 0, stream>>>(Wk, wtk);
    wconv_kernel<<<NDD / 256, 256, 0, stream>>>(Wv, wtv);
    wconv_kernel<<<NDD / 256, 256, 0, stream>>>(Wo, wto);
    gemm_qkv_kernel<<<dim3(512, 3), 256, 0, stream>>>(hbf, wtq, wtk, wtv,
                                                      qbf, kbf, vbf);
    vtrans_kernel<<<NBTD / 256, 256, 0, stream>>>(vbf, vtb);
    attn_kernel<<<512, 256, 0, stream>>>(qbf, kbf, vtb, obf);
    gemm_out_kernel<<<512, 256, 0, stream>>>(obf, wto, x, bo, out);
}
